// MultiHeadAttention_12412455485989
// MI455X (gfx1250) — compile-verified, matrix operand order checked by reading, not run on hardware
//
#include <hip/hip_runtime.h>

// Problem constants (match reference)
#define B_  4
#define S_  1500
#define NS  1024
#define H_  16
#define D_  64
#define M_  (B_ * S_)   // 6000 rows for the projection GEMMs

typedef __attribute__((ext_vector_type(16))) __bf16 v16bf;
typedef __attribute__((ext_vector_type(8)))  float  v8f;
typedef __attribute__((ext_vector_type(4)))  int    v4i;

union FragBF { v16bf v; unsigned short u[16]; };

__device__ __forceinline__ unsigned short f2bf(float f) {
  unsigned int u = __float_as_uint(f);
  u += 0x7FFFu + ((u >> 16) & 1u);      // round-to-nearest-even
  return (unsigned short)(u >> 16);
}

// ---- CDNA5 async global->LDS path (gated; sync fallback otherwise) --------
#if defined(__gfx1250__) || defined(__HIP_DEVICE_COMPILE__)
#if __has_builtin(__builtin_amdgcn_global_load_async_to_lds_b128) && \
    __has_builtin(__builtin_amdgcn_s_wait_asynccnt)
#define USE_ASYNC 1
#endif
#endif
#ifndef USE_ASYNC
#define USE_ASYNC 0
#endif

#if USE_ASYNC
#define NBUF 2
typedef __attribute__((address_space(1))) v4i* gp_v4i;   // global, typed
typedef __attribute__((address_space(3))) v4i* lp_v4i;   // LDS, typed
#define ASYNC_B128(dst_lds, src_glb)                                   \
  __builtin_amdgcn_global_load_async_to_lds_b128(                      \
      (gp_v4i)(src_glb), (lp_v4i)(dst_lds), 0, 0)
#else
#define NBUF 1
#endif

// ---------------------------------------------------------------------------
// Elementwise fp32 -> bf16 (x and the four weight matrices), vectorized x4.
// ---------------------------------------------------------------------------
__global__ __launch_bounds__(256)
void f32_to_bf16(const float* __restrict__ src, unsigned short* __restrict__ dst,
                 int n4)
{
  int i = blockIdx.x * 256 + threadIdx.x;
  if (i < n4) {
    float4 f = ((const float4*)src)[i];
    ushort4 o;
    o.x = f2bf(f.x); o.y = f2bf(f.y); o.z = f2bf(f.z); o.w = f2bf(f.w);
    ((ushort4*)dst)[i] = o;
  }
}

// ---------------------------------------------------------------------------
// GEMM:  out[m][n] = sum_k A[m][k] * W[n][k] + bias[n]   (A @ W^T + b), bf16 in.
// Block tile 128x128, 8 waves each 64x32 (4x2 WMMA tiles), BK=32.
// Async double-buffered LDS staging when available.
// ---------------------------------------------------------------------------
__global__ __launch_bounds__(256)
void gemm_bf16(const unsigned short* __restrict__ Ab,
               const unsigned short* __restrict__ Wb,
               const float* __restrict__ bias,
               float* __restrict__ outF, unsigned short* __restrict__ outB,
               int M, int N, int K)
{
  __shared__ __align__(16) unsigned short As[NBUF][128][40]; // [m][k], 80B rows
  __shared__ __align__(16) unsigned short Bs[NBUF][128][40]; // [n][k]

  const int tid  = threadIdx.x;
  const int wave = tid >> 5;
  const int lane = tid & 31;
  const int lr   = lane & 15;
  const int hi   = lane >> 4;
  const int klo  = hi * 8;
  const int waveM = wave >> 2, waveN = wave & 3;   // 2 x 4 wave grid
  const int m0 = blockIdx.x * 128, n0 = blockIdx.y * 128;
  const int mW = waveM * 64, nW = waveN * 32;
  const int nK = K >> 5;

  v8f c[4][2];
#pragma unroll
  for (int tm = 0; tm < 4; ++tm)
#pragma unroll
    for (int tn = 0; tn < 2; ++tn)
      c[tm][tn] = (v8f){0.f, 0.f, 0.f, 0.f, 0.f, 0.f, 0.f, 0.f};

#if USE_ASYNC
  auto stage = [&](int buf, int kb) {
    const int k0 = kb << 5;
#pragma unroll
    for (int i = 0; i < 2; ++i) {
      int g = i * 256 + tid;                // 0..511 groups of 8 bf16 (16B)
      int row = g >> 2, c8 = (g & 3) * 8;
      int m = m0 + row; if (m > M - 1) m = M - 1;   // clamp: stay in-tensor
      ASYNC_B128(&As[buf][row][c8], Ab + (size_t)m * K + k0 + c8);
      ASYNC_B128(&Bs[buf][row][c8], Wb + (size_t)(n0 + row) * K + k0 + c8);
    }
  };
  stage(0, 0);
  if (nK > 1) stage(1, 1);
#endif

  for (int kb = 0; kb < nK; ++kb) {
    const int cur = kb & (NBUF - 1);
#if USE_ASYNC
    if (kb + 1 < nK) __builtin_amdgcn_s_wait_asynccnt(4);  // tile kb done
    else             __builtin_amdgcn_s_wait_asynccnt(0);
    __syncthreads();
#else
    const int k0 = kb << 5;
#pragma unroll
    for (int i = 0; i < 16; ++i) {
      int idx = i * 256 + tid;
      int r = idx >> 5, kk = idx & 31;
      int m = m0 + r; if (m > M - 1) m = M - 1;
      As[0][r][kk] = Ab[(size_t)m * K + k0 + kk];
      Bs[0][r][kk] = Wb[(size_t)(n0 + r) * K + k0 + kk];
    }
    __syncthreads();
#endif

    FragBF a[4], b[2];
#pragma unroll
    for (int tm = 0; tm < 4; ++tm) {
      int row = mW + tm * 16 + lr;
#pragma unroll
      for (int j = 0; j < 16; ++j)
        a[tm].u[j] = As[cur][row][klo + (j < 8 ? j : j + 8)];
    }
#pragma unroll
    for (int tn = 0; tn < 2; ++tn) {
      int col = nW + tn * 16 + lr;
#pragma unroll
      for (int j = 0; j < 16; ++j)
        b[tn].u[j] = Bs[cur][col][klo + (j < 8 ? j : j + 8)];
    }

#pragma unroll
    for (int tm = 0; tm < 4; ++tm)
#pragma unroll
      for (int tn = 0; tn < 2; ++tn)
        c[tm][tn] = __builtin_amdgcn_wmma_f32_16x16x32_bf16(
            false, a[tm].v, false, b[tn].v, (short)0, c[tm][tn], false, false);

    __syncthreads();
#if USE_ASYNC
    if (kb + 2 < nK) stage(cur, kb + 2);
#endif
  }

  // Epilogue: bias add, fp32 and/or bf16 store.
#pragma unroll
  for (int tm = 0; tm < 4; ++tm) {
#pragma unroll
    for (int tn = 0; tn < 2; ++tn) {
      int n = n0 + nW + tn * 16 + lr;
      float bb = bias ? bias[n] : 0.f;
#pragma unroll
      for (int r = 0; r < 8; ++r) {
        int m = m0 + mW + tm * 16 + r + 8 * hi;
        if (m < M) {
          float val = c[tm][tn][r] + bb;
          if (outF) outF[(size_t)m * N + n] = val;
          if (outB) outB[(size_t)m * N + n] = f2bf(val);
        }
      }
    }
  }
}

// ---------------------------------------------------------------------------
// Fused attention per (b,h): async-staged 64-wide K/V tiles; masked scaled
// scores streamed to qk output; online (flash) softmax; P @ V in fp32.
// Block: 256 threads = 8 waves; each wave owns 16 query rows (128/block).
// ---------------------------------------------------------------------------
__global__ __launch_bounds__(256)
void attn_fused(const unsigned short* __restrict__ qw,
                const unsigned short* __restrict__ kw,
                const unsigned short* __restrict__ vw,
                unsigned short* __restrict__ wvw,
                float* __restrict__ qk_out)
{
  __shared__ __align__(16) unsigned short Kn[NBUF][64][72]; // [s][d] natural
  __shared__ __align__(16) unsigned short Vs[NBUF][64][72]; // [s][d] natural
  __shared__ __align__(16) unsigned short Ps[128][72];      // wave-private rows

  const int tid  = threadIdx.x;
  const int wave = tid >> 5;
  const int lane = tid & 31;
  const int lr   = lane & 15;
  const int hi   = lane >> 4;
  const int klo  = hi * 8;
  const int bh = blockIdx.y;
  const int b = bh >> 4, h = bh & 15;
  const int qBase = blockIdx.x * 128;
  const float scale = 0.125f;             // d^-0.5, d = 64
  const int KT = (S_ + 63) / 64;          // 24 k-tiles

  // Q fragments (this wave's 16 rows), resident for the whole pass.
  FragBF qf[2];
  {
    int row = qBase + wave * 16 + lr;
    if (row < S_) {
      const unsigned short* qp = qw + ((size_t)(b * S_ + row)) * NS + h * D_;
#pragma unroll
      for (int ks = 0; ks < 2; ++ks)
#pragma unroll
        for (int j = 0; j < 16; ++j)
          qf[ks].u[j] = qp[ks * 32 + klo + (j < 8 ? j : j + 8)];
    } else {
#pragma unroll
      for (int ks = 0; ks < 2; ++ks)
#pragma unroll
        for (int j = 0; j < 16; ++j)
          qf[ks].u[j] = 0;
    }
  }

  v8f o_acc[4];
#pragma unroll
  for (int t = 0; t < 4; ++t)
    o_acc[t] = (v8f){0.f, 0.f, 0.f, 0.f, 0.f, 0.f, 0.f, 0.f};
  float runmax[8], runsum[8];
#pragma unroll
  for (int r = 0; r < 8; ++r) { runmax[r] = -3.0e38f; runsum[r] = 0.f; }

#if USE_ASYNC
  auto stageKV = [&](int buf, int kt) {
    const int kBase = kt * 64;
#pragma unroll
    for (int i = 0; i < 2; ++i) {
      int g = i * 256 + tid;               // 0..511 groups of 8 bf16 (16B)
      int sl = g >> 3, c8 = (g & 7) * 8;
      int s = kBase + sl; if (s > S_ - 1) s = S_ - 1;  // clamp: in-tensor
      size_t base = ((size_t)(b * S_ + s)) * NS + h * D_ + c8;
      ASYNC_B128(&Kn[buf][sl][c8], kw + base);
      ASYNC_B128(&Vs[buf][sl][c8], vw + base);
    }
  };
  stageKV(0, 0);
  stageKV(1, 1);
#endif

  for (int kt = 0; kt < KT; ++kt) {
    const int cur = kt & (NBUF - 1);
    const int kBase = kt * 64;
#if USE_ASYNC
    if (kt + 1 < KT) __builtin_amdgcn_s_wait_asynccnt(4);
    else             __builtin_amdgcn_s_wait_asynccnt(0);
    __syncthreads();
#else
#pragma unroll
    for (int i = 0; i < 16; ++i) {
      int idx = i * 256 + tid;
      int sl = idx >> 6, dd = idx & 63;
      int s = kBase + sl; if (s > S_ - 1) s = S_ - 1;
      size_t base = ((size_t)(b * S_ + s)) * NS + h * D_ + dd;
      Kn[0][sl][dd] = kw[base];
      Vs[0][sl][dd] = vw[base];
    }
    __syncthreads();
#endif

    // Scores: 4 column tiles x 2 K-steps of WMMA.  scores = Q @ K^T
    v8f sc[4];
#pragma unroll
    for (int tn = 0; tn < 4; ++tn) {
      sc[tn] = (v8f){0.f, 0.f, 0.f, 0.f, 0.f, 0.f, 0.f, 0.f};
#pragma unroll
      for (int ks = 0; ks < 2; ++ks) {
        FragBF bk;
        int col = tn * 16 + lr;
#pragma unroll
        for (int j = 0; j < 16; ++j)
          bk.u[j] = Kn[cur][col][ks * 32 + klo + (j < 8 ? j : j + 8)];
        sc[tn] = __builtin_amdgcn_wmma_f32_16x16x32_bf16(
            false, qf[ks].v, false, bk.v, (short)0, sc[tn], false, false);
      }
    }

    // Scale + causal mask, stream qk out, online-softmax update, stage P.
#pragma unroll
    for (int r = 0; r < 8; ++r) {
      const int qr = qBase + wave * 16 + r + 8 * hi;
      float m = -3.0e38f;
#pragma unroll
      for (int tn = 0; tn < 4; ++tn) {
        int kc = kBase + tn * 16 + lr;
        float val = sc[tn][r] * scale;
        if (kc > qr) val -= 1.0e4f;                      // reference mask value
        float sval = ((qr < S_) && (kc < S_)) ? val : -3.0e38f;
        m = fmaxf(m, sval);
      }
      m = fmaxf(m, __shfl_xor(m, 1, 32));
      m = fmaxf(m, __shfl_xor(m, 2, 32));
      m = fmaxf(m, __shfl_xor(m, 4, 32));
      m = fmaxf(m, __shfl_xor(m, 8, 32));
      float nm    = fmaxf(runmax[r], m);
      float alpha = __expf(runmax[r] - nm);
      runmax[r] = nm;

      float ps = 0.f;
      const int prow = wave * 16 + r + 8 * hi;
#pragma unroll
      for (int tn = 0; tn < 4; ++tn) {
        int kc = kBase + tn * 16 + lr;
        float val = sc[tn][r] * scale;
        if (kc > qr) val -= 1.0e4f;
        bool inb = (qr < S_) && (kc < S_);
        if (inb)
          qk_out[(((size_t)(b * H_ + h)) * S_ + qr) * S_ + kc] = val;
        float sval = inb ? val : -3.0e38f;
        float p = __expf(sval - nm);
        ps += p;
        Ps[prow][tn * 16 + lr] = f2bf(p);                // wave-private rows
      }
      ps += __shfl_xor(ps, 1, 32);
      ps += __shfl_xor(ps, 2, 32);
      ps += __shfl_xor(ps, 4, 32);
      ps += __shfl_xor(ps, 8, 32);
      runsum[r] = runsum[r] * alpha + ps;
#pragma unroll
      for (int t = 0; t < 4; ++t)
        o_acc[t][r] *= alpha;
    }

    // O += P @ V   (K-dim = 64 key positions, 2 WMMA steps; 4 d-tiles)
    FragBF pa[2];
#pragma unroll
    for (int ks = 0; ks < 2; ++ks) {
      int prow = wave * 16 + lr;
#pragma unroll
      for (int j = 0; j < 16; ++j)
        pa[ks].u[j] = Ps[prow][ks * 32 + klo + (j < 8 ? j : j + 8)];
    }
#pragma unroll
    for (int td = 0; td < 4; ++td) {
#pragma unroll
      for (int ks = 0; ks < 2; ++ks) {
        FragBF bv;
        int col = td * 16 + lr;
#pragma unroll
        for (int j = 0; j < 16; ++j)                     // strided (V natural)
          bv.u[j] = Vs[cur][ks * 32 + klo + (j < 8 ? j : j + 8)][col];
        o_acc[td] = __builtin_amdgcn_wmma_f32_16x16x32_bf16(
            false, pa[ks].v, false, bv.v, (short)0, o_acc[td], false, false);
      }
    }

    __syncthreads();
#if USE_ASYNC
    if (kt + 2 < KT) stageKV(cur, kt + 2);
#endif
  }

  // Normalize and write wv (bf16 workspace, [B,S,H,D] == [B,S,N] layout).
#pragma unroll
  for (int td = 0; td < 4; ++td) {
    int dd = td * 16 + lr;
#pragma unroll
    for (int r = 0; r < 8; ++r) {
      int qr = qBase + wave * 16 + r + 8 * hi;
      if (qr < S_) {
        float val = o_acc[td][r] / runsum[r];
        wvw[((size_t)(b * S_ + qr)) * NS + h * D_ + dd] = f2bf(val);
      }
    }
  }
}

// ---------------------------------------------------------------------------
extern "C" void kernel_launch(void* const* d_in, const int* in_sizes, int n_in,
                              void* d_out, int out_size, void* d_ws, size_t ws_size,
                              hipStream_t stream)
{
  (void)in_sizes; (void)n_in; (void)out_size; (void)ws_size;
  const float* x  = (const float*)d_in[0];
  // d_in[1] = qk_mask: recomputed analytically (triu k=1 -> -1e4)
  const float* Wq = (const float*)d_in[2];
  const float* bq = (const float*)d_in[3];
  const float* Wk = (const float*)d_in[4];
  const float* Wv = (const float*)d_in[5];
  const float* bv = (const float*)d_in[6];
  const float* Wo = (const float*)d_in[7];
  const float* bo = (const float*)d_in[8];

  // d_out layout: out | qk | new_k | new_v
  float* out = (float*)d_out;
  float* qk  = out + (size_t)M_ * NS;
  float* kO  = qk  + (size_t)B_ * H_ * S_ * S_;
  float* vO  = kO  + (size_t)M_ * NS;

  // workspace (bf16): q | k | v | wv | x | Wq | Wk | Wv | Wo
  const size_t nT = (size_t)M_ * NS;        // 6.144M
  const size_t nW = (size_t)NS * NS;        // 1.048576M
  unsigned short* qW  = (unsigned short*)d_ws;
  unsigned short* kW  = qW  + nT;
  unsigned short* vW  = kW  + nT;
  unsigned short* wvW = vW  + nT;
  unsigned short* xB  = wvW + nT;
  unsigned short* WqB = xB  + nT;
  unsigned short* WkB = WqB + nW;
  unsigned short* WvB = WkB + nW;
  unsigned short* WoB = WvB + nW;

  dim3 blk(256);
  // 1) one-shot fp32 -> bf16 conversions
  f32_to_bf16<<<dim3((nT / 4 + 255) / 256), blk, 0, stream>>>(x,  xB,  (int)(nT / 4));
  f32_to_bf16<<<dim3((nW / 4 + 255) / 256), blk, 0, stream>>>(Wq, WqB, (int)(nW / 4));
  f32_to_bf16<<<dim3((nW / 4 + 255) / 256), blk, 0, stream>>>(Wk, WkB, (int)(nW / 4));
  f32_to_bf16<<<dim3((nW / 4 + 255) / 256), blk, 0, stream>>>(Wv, WvB, (int)(nW / 4));
  f32_to_bf16<<<dim3((nW / 4 + 255) / 256), blk, 0, stream>>>(Wo, WoB, (int)(nW / 4));

  // 2) Q/K/V projections (K and V also land in d_out as fp32)
  dim3 g1((M_ + 127) / 128, NS / 128);      // 47 x 8
  gemm_bf16<<<g1, blk, 0, stream>>>(xB, WqB, bq,      nullptr, qW, M_, NS, NS);
  gemm_bf16<<<g1, blk, 0, stream>>>(xB, WkB, nullptr, kO,      kW, M_, NS, NS);
  gemm_bf16<<<g1, blk, 0, stream>>>(xB, WvB, bv,      vO,      vW, M_, NS, NS);

  // 3) fused masked-softmax attention (streams qk tensor)
  dim3 g2((S_ + 127) / 128, B_ * H_);       // 12 x 64
  attn_fused<<<g2, blk, 0, stream>>>(qW, kW, vW, wvW, qk);

  // 4) output projection
  gemm_bf16<<<g1, blk, 0, stream>>>(wvW, WoB, bo, out, nullptr, M_, NS, NS);
}
